// HybridGCNGATModel_8770323218998
// MI455X (gfx1250) — matrix operand order repaired and checked by reading, not compiled, not run on hardware
//
#include <hip/hip_runtime.h>
#include <math.h>

#define NN 50000
#define EE 800000
#define DD 128
#define HH 8
#define CC 16
#define LN_EPS 1e-5f
#define NEG_SLOPE 0.2f

typedef __attribute__((ext_vector_type(16))) __bf16 v16bf;
typedef __attribute__((ext_vector_type(8)))  __bf16 v8bf;
typedef __attribute__((ext_vector_type(4)))  __bf16 v4bf;
typedef __attribute__((ext_vector_type(8)))  float  v8f;

// ---------------- helpers ----------------
static __device__ inline __bf16 f2bf(float f) {
    unsigned u = __float_as_uint(f);
    unsigned r = (u + 0x7fffu + ((u >> 16) & 1u)) >> 16;   // RNE truncate
    unsigned short s = (unsigned short)r;
    return __builtin_bit_cast(__bf16, s);
}
static __device__ inline int f2o(float f) {              // order-preserving float->int
    int i = __float_as_int(f);
    return i >= 0 ? i : (i ^ 0x7FFFFFFF);
}
static __device__ inline float o2f(int o) {
    int i = o >= 0 ? o : (o ^ 0x7FFFFFFF);
    return __int_as_float(i);
}

// ---------------- conversion / packing ----------------
// Elementwise f32 -> bf16, 4 elements per thread (vector load + 8B store).
__global__ void f32_to_bf16(const float* __restrict__ x, __bf16* __restrict__ y, long n) {
    long i = ((long)blockIdx.x * blockDim.x + threadIdx.x) * 4;
    if (i >= n) return;
    const float4 v = *(const float4*)(x + i);
    v4bf o; o[0] = f2bf(v.x); o[1] = f2bf(v.y); o[2] = f2bf(v.z); o[3] = f2bf(v.w);
    *(v4bf*)(y + i) = o;
}

// Pack 128x128 f32 weight (row-major [K][N]) into WMMA B-fragment order (bf16):
// flat = ((nt*4 + kbi)*32 + lane)*16 + i  maps to  W[kbi*32 + (i&7) + 8*(lane>>4) + 16*(i>>3)][nt*16 + (lane&15)]
__global__ void pack_w(const float* __restrict__ W, __bf16* __restrict__ Wpk) {
    int t = blockIdx.x * blockDim.x + threadIdx.x;
    if (t >= DD * DD) return;
    const int i    = t & 15;
    const int lane = (t >> 4) & 31;
    const int kbi  = (t >> 9) & 3;
    const int nt   = t >> 11;
    const int k = kbi * 32 + (i & 7) + 8 * (lane >> 4) + 16 * (i >> 3);
    const int n = nt * 16 + (lane & 15);
    Wpk[t] = f2bf(W[k * DD + n]);
}

// ---------------- GEMM: out[M x 128] = A[M x 128] @ W[128 x 128] + bias ----------------
// A pre-converted to bf16 row-major; W pre-packed into fragment order.
// One wave = one 16x16 tile; 4 fully-unrolled v_wmma_f32_16x16x32_bf16.
__global__ void gemm128_bf16_wmma(const __bf16* __restrict__ Abf,
                                  const __bf16* __restrict__ Wpk,
                                  const float* __restrict__ bias,
                                  float* __restrict__ out, int M) {
    const int lane = threadIdx.x & 31;
    const int wave = threadIdx.x >> 5;
    const int wpb  = blockDim.x >> 5;
    const int tile = blockIdx.x * wpb + wave;
    const int ntc  = DD / 16;                  // 8 column tiles
    const int mt   = tile / ntc;
    const int nt   = tile % ntc;
    if (mt * 16 >= M) return;                  // uniform per wave: EXEC stays all-ones

    const int kgrp = lane >> 4;                // 0 or 1
    const int lrow = lane & 15;
    const int rowA = mt * 16 + lrow;
    const int colB = nt * 16 + lrow;

    v8f c;
    const float bv = bias ? bias[colB] : 0.0f;
#pragma unroll
    for (int r = 0; r < 8; ++r) c[r] = bv;

    const __bf16* __restrict__ arow = Abf + (long)rowA * DD;
    const __bf16* __restrict__ bp   = Wpk + (nt * 2048 + lane * 16);

#pragma unroll
    for (int kbi = 0; kbi < 4; ++kbi) {
        const int kb = kbi * 32;
        // A 16x32 bf16 fragment: lanes 0-15 -> K {kb+0..7, kb+16..23}; lanes 16-31 -> +8
        const v8bf alo = *(const v8bf*)(arow + kb + 8 * kgrp);
        const v8bf ahi = *(const v8bf*)(arow + kb + 16 + 8 * kgrp);
        v16bf a;
#pragma unroll
        for (int i = 0; i < 8; ++i) { a[i] = alo[i]; a[i + 8] = ahi[i]; }
        // B fragment: 32 contiguous bytes per lane from the packed weight
        const v16bf b = *(const v16bf*)(bp + kbi * 512);
        c = __builtin_amdgcn_wmma_f32_16x16x32_bf16(
                /*neg_a=*/false, a, /*neg_b=*/false, b,
                /*c_mod=*/(short)0, c, /*reuse_a=*/false, /*reuse_b=*/false);
    }
    const int rowBase = mt * 16 + 8 * kgrp;
#pragma unroll
    for (int r = 0; r < 8; ++r)
        out[(long)(rowBase + r) * DD + colB] = c[r];
}

// ---------------- degrees ----------------
__global__ void deg_init(float* deg, int n) {
    int i = blockIdx.x * blockDim.x + threadIdx.x;
    if (i < n) deg[i] = 1.0f;                  // self loop
}
__global__ void deg_accum(const int* __restrict__ dst, float* deg, int e) {
    int i = blockIdx.x * blockDim.x + threadIdx.x;
    if (i < e) atomicAdd(&deg[dst[i]], 1.0f);
}
__global__ void deg_to_dinv(float* deg, int n) {
    int i = blockIdx.x * blockDim.x + threadIdx.x;
    if (i < n) deg[i] = rsqrtf(deg[i]);        // deg >= 1 always
}

// ---------------- GCN scatter: out[d] += xw[s] * dinv[s]*dinv[d] (one wave/edge) ----------------
__global__ void gcn_scatter(const float* __restrict__ xw,
                            const int* __restrict__ src, const int* __restrict__ dst,
                            const float* __restrict__ dinv, float* __restrict__ out,
                            int e_cnt, int n_cnt) {
    const int e    = blockIdx.x * (blockDim.x >> 5) + (threadIdx.x >> 5);
    const int lane = threadIdx.x & 31;
    if (e >= e_cnt + n_cnt) return;
    int s, d;
    if (e < e_cnt) { s = src[e]; d = dst[e]; } else { s = d = e - e_cnt; }
    const float norm = dinv[s] * dinv[d];
    const float4 v = ((const float4*)(xw + (long)s * DD))[lane];
    float* od = out + (long)d * DD + lane * 4;
    atomicAdd(od + 0, v.x * norm);
    atomicAdd(od + 1, v.y * norm);
    atomicAdd(od + 2, v.z * norm);
    atomicAdd(od + 3, v.w * norm);
}

// ---------------- graph LayerNorm: stats then apply ----------------
__global__ void ln_stats(const float* __restrict__ x, const float* __restrict__ bias,
                         float* stats, long n) {
    float s = 0.f, s2 = 0.f;
    for (long i = (long)blockIdx.x * blockDim.x + threadIdx.x; i < n;
         i += (long)gridDim.x * blockDim.x) {
        float v = x[i] + (bias ? bias[i & (DD - 1)] : 0.f);
        s += v; s2 += v * v;
    }
    __shared__ float shs[256], sh2[256];
    const int t = threadIdx.x;
    shs[t] = s; sh2[t] = s2;
    __syncthreads();
    for (int off = 128; off > 0; off >>= 1) {
        if (t < off) { shs[t] += shs[t + off]; sh2[t] += sh2[t + off]; }
        __syncthreads();
    }
    if (t == 0) { atomicAdd(&stats[0], shs[0]); atomicAdd(&stats[1], sh2[0]); }
}
__global__ void ln_apply(float* __restrict__ x, const float* __restrict__ bias,
                         const float* __restrict__ g, const float* __restrict__ b,
                         const float* __restrict__ stats, long n, int relu) {
    const long i = (long)blockIdx.x * blockDim.x + threadIdx.x;
    if (i >= n) return;
    const float invn  = 1.0f / (float)n;
    const float mean  = stats[0] * invn;
    float var = stats[1] * invn - mean * mean;
    var = var > 0.f ? var : 0.f;
    const float denom = sqrtf(var) + LN_EPS;
    const int c = (int)(i & (DD - 1));
    float v = x[i] + (bias ? bias[c] : 0.f);
    v = (v - mean) / denom * g[c] + b[c];
    if (relu) v = fmaxf(v, 0.f);
    x[i] = v;
}

// ---------------- GATv2 ----------------
__global__ void smax_init(int* smax, int n) {
    int i = blockIdx.x * blockDim.x + threadIdx.x;
    if (i < n) smax[i] = f2o(-INFINITY);
}
__global__ void gat_score(const float* __restrict__ xl, const float* __restrict__ xr,
                          const float* __restrict__ att,
                          const int* __restrict__ src, const int* __restrict__ dst,
                          float* __restrict__ scores, int* __restrict__ smax,
                          int e_cnt, int n_cnt, int heads, int ch) {
    const long t = (long)blockIdx.x * blockDim.x + threadIdx.x;
    const long total = (long)(e_cnt + n_cnt) * heads;
    if (t >= total) return;
    const int  h = (int)(t % heads);
    const long e = t / heads;
    int s, d;
    if (e < e_cnt) { s = src[e]; d = dst[e]; } else { s = d = (int)(e - e_cnt); }
    const float* pl = xl + (long)s * DD + h * ch;
    const float* pr = xr + (long)d * DD + h * ch;
    const float* pa = att + h * ch;
    float sc = 0.f;
    for (int c = 0; c < ch; ++c) {
        float v = pl[c] + pr[c];
        v = v > 0.f ? v : v * NEG_SLOPE;       // leaky_relu, slope 0.2
        sc += v * pa[c];
    }
    scores[t] = sc;
    atomicMax(&smax[(long)d * heads + h], f2o(sc));
}
__global__ void gat_expsum(float* __restrict__ scores, const int* __restrict__ dst,
                           const int* __restrict__ smax, float* __restrict__ denom,
                           int e_cnt, int n_cnt, int heads) {
    const long t = (long)blockIdx.x * blockDim.x + threadIdx.x;
    const long total = (long)(e_cnt + n_cnt) * heads;
    if (t >= total) return;
    const int  h = (int)(t % heads);
    const long e = t / heads;
    const int  d = (e < e_cnt) ? dst[e] : (int)(e - e_cnt);
    const float a = expf(scores[t] - o2f(smax[(long)d * heads + h]));
    scores[t] = a;
    atomicAdd(&denom[(long)d * heads + h], a);
}
__global__ void gat_agg(const float* __restrict__ xl, const float* __restrict__ scores,
                        const float* __restrict__ denom,
                        const int* __restrict__ src, const int* __restrict__ dst,
                        float* __restrict__ out, int e_cnt, int n_cnt, int heads, int ch) {
    const int e    = blockIdx.x * (blockDim.x >> 5) + (threadIdx.x >> 5);
    const int lane = threadIdx.x & 31;
    if (e >= e_cnt + n_cnt) return;
    int s, d;
    if (e < e_cnt) { s = src[e]; d = dst[e]; } else { s = d = e - e_cnt; }
    const float* ps = xl + (long)s * DD;
    float*       po = out + (long)d * DD;
#pragma unroll
    for (int q = 0; q < 4; ++q) {
        const int j = lane * 4 + q;
        const int h = j / ch;
        const float alpha = scores[(long)e * heads + h] / denom[(long)d * heads + h];
        atomicAdd(&po[j], alpha * ps[j]);
    }
}

// ---------------- host orchestration ----------------
extern "C" void kernel_launch(void* const* d_in, const int* in_sizes, int n_in,
                              void* d_out, int out_size, void* d_ws, size_t ws_size,
                              hipStream_t stream) {
    const float* x    = (const float*)d_in[0];
    const int*   ei   = (const int*)d_in[1];
    const int*   srcP = ei;
    const int*   dstP = ei + EE;
    const float *W0 = (const float*)d_in[2],  *b0  = (const float*)d_in[3];
    const float *g0 = (const float*)d_in[4],  *be0 = (const float*)d_in[5];
    const float *Wl1= (const float*)d_in[6],  *bl1 = (const float*)d_in[7];
    const float *Wr1= (const float*)d_in[8],  *br1 = (const float*)d_in[9];
    const float *at1= (const float*)d_in[10], *bg1 = (const float*)d_in[11];
    const float *g1 = (const float*)d_in[12], *be1 = (const float*)d_in[13];
    const float *W2 = (const float*)d_in[14], *b2  = (const float*)d_in[15];
    const float *g2 = (const float*)d_in[16], *be2 = (const float*)d_in[17];
    const float *Wl3= (const float*)d_in[18], *bl3 = (const float*)d_in[19];
    const float *Wr3= (const float*)d_in[20], *br3 = (const float*)d_in[21];
    const float *at3= (const float*)d_in[22], *bg3 = (const float*)d_in[23];
    const float *g3 = (const float*)d_in[24], *be3 = (const float*)d_in[25];
    float* out = (float*)d_out;

    // workspace bump allocation (all section sizes are multiples of 64B)
    const size_t ND = (size_t)NN * DD;
    float* buf0   = (float*)d_ws;
    float* buf1   = buf0 + ND;
    float* xlB    = buf1 + ND;
    float* xrB    = xlB + ND;
    float* scores = xrB + ND;                         // (E+N)*H floats max
    float* smaxF  = scores + (size_t)(EE + NN) * HH;  // used as int*
    float* denom  = smaxF + (size_t)NN * HH;
    float* dinv   = denom + (size_t)NN * HH;
    float* stats  = dinv + NN;                        // 2 floats (padded to 16)
    __bf16* Abf   = (__bf16*)(stats + 16);            // ND bf16
    __bf16* wpk   = Abf + ND;                         // 6 * 128*128 bf16
    __bf16* wpkW0  = wpk;
    __bf16* wpkWl1 = wpk + 1 * DD * DD;
    __bf16* wpkWr1 = wpk + 2 * DD * DD;
    __bf16* wpkW2  = wpk + 3 * DD * DD;
    __bf16* wpkWl3 = wpk + 4 * DD * DD;
    __bf16* wpkWr3 = wpk + 5 * DD * DD;

    const int TPB = 256;
    const int gemmBlocks = ((NN / 16) * (DD / 16) + 7) / 8;        // 8 waves/block
    const int edgeWaveBl = (EE + NN + 7) / 8;                      // one wave/edge
    const int cvtBlocks  = (int)((ND / 4 + TPB - 1) / TPB);
    const int packBlocks = (DD * DD + TPB - 1) / TPB;
    const long totEH8 = (long)(EE + NN) * HH;
    const long totEH1 = (long)(EE + NN);

    // --- pack all constant weights once ---
    pack_w<<<packBlocks, TPB, 0, stream>>>(W0,  wpkW0);
    pack_w<<<packBlocks, TPB, 0, stream>>>(Wl1, wpkWl1);
    pack_w<<<packBlocks, TPB, 0, stream>>>(Wr1, wpkWr1);
    pack_w<<<packBlocks, TPB, 0, stream>>>(W2,  wpkW2);
    pack_w<<<packBlocks, TPB, 0, stream>>>(Wl3, wpkWl3);
    pack_w<<<packBlocks, TPB, 0, stream>>>(Wr3, wpkWr3);

    // --- degrees -> dinv ---
    deg_init   <<<(NN + TPB - 1) / TPB, TPB, 0, stream>>>(dinv, NN);
    deg_accum  <<<(EE + TPB - 1) / TPB, TPB, 0, stream>>>(dstP, dinv, EE);
    deg_to_dinv<<<(NN + TPB - 1) / TPB, TPB, 0, stream>>>(dinv, NN);

    // ===== Layer 0: GCN + graphLN + relu  (h -> buf1) =====
    f32_to_bf16<<<cvtBlocks, TPB, 0, stream>>>(x, Abf, (long)ND);
    gemm128_bf16_wmma<<<gemmBlocks, TPB, 0, stream>>>(Abf, wpkW0, b0, buf0, NN);
    hipMemsetAsync(buf1, 0, ND * sizeof(float), stream);
    gcn_scatter<<<edgeWaveBl, TPB, 0, stream>>>(buf0, srcP, dstP, dinv, buf1, EE, NN);
    hipMemsetAsync(stats, 0, 2 * sizeof(float), stream);
    ln_stats<<<2048, TPB, 0, stream>>>(buf1, nullptr, stats, (long)ND);
    ln_apply<<<(int)((ND + TPB - 1) / TPB), TPB, 0, stream>>>(buf1, nullptr, g0, be0, stats, (long)ND, 1);

    // ===== Layer 1: GATv2 (H=8,C=16) + graphLN + relu  (h -> buf0) =====
    f32_to_bf16<<<cvtBlocks, TPB, 0, stream>>>(buf1, Abf, (long)ND);
    gemm128_bf16_wmma<<<gemmBlocks, TPB, 0, stream>>>(Abf, wpkWl1, bl1, xlB, NN);
    gemm128_bf16_wmma<<<gemmBlocks, TPB, 0, stream>>>(Abf, wpkWr1, br1, xrB, NN);
    smax_init<<<(NN * HH + TPB - 1) / TPB, TPB, 0, stream>>>((int*)smaxF, NN * HH);
    hipMemsetAsync(denom, 0, (size_t)NN * HH * sizeof(float), stream);
    gat_score <<<(int)((totEH8 + TPB - 1) / TPB), TPB, 0, stream>>>(xlB, xrB, at1, srcP, dstP,
                                                                    scores, (int*)smaxF, EE, NN, HH, CC);
    gat_expsum<<<(int)((totEH8 + TPB - 1) / TPB), TPB, 0, stream>>>(scores, dstP, (int*)smaxF,
                                                                    denom, EE, NN, HH);
    hipMemsetAsync(buf0, 0, ND * sizeof(float), stream);
    gat_agg<<<edgeWaveBl, TPB, 0, stream>>>(xlB, scores, denom, srcP, dstP, buf0, EE, NN, HH, CC);
    hipMemsetAsync(stats, 0, 2 * sizeof(float), stream);
    ln_stats<<<2048, TPB, 0, stream>>>(buf0, bg1, stats, (long)ND);
    ln_apply<<<(int)((ND + TPB - 1) / TPB), TPB, 0, stream>>>(buf0, bg1, g1, be1, stats, (long)ND, 1);

    // ===== Layer 2: GCN + graphLN + relu  (h -> buf1) =====
    f32_to_bf16<<<cvtBlocks, TPB, 0, stream>>>(buf0, Abf, (long)ND);
    gemm128_bf16_wmma<<<gemmBlocks, TPB, 0, stream>>>(Abf, wpkW2, b2, xlB, NN);
    hipMemsetAsync(buf1, 0, ND * sizeof(float), stream);
    gcn_scatter<<<edgeWaveBl, TPB, 0, stream>>>(xlB, srcP, dstP, dinv, buf1, EE, NN);
    hipMemsetAsync(stats, 0, 2 * sizeof(float), stream);
    ln_stats<<<2048, TPB, 0, stream>>>(buf1, nullptr, stats, (long)ND);
    ln_apply<<<(int)((ND + TPB - 1) / TPB), TPB, 0, stream>>>(buf1, nullptr, g2, be2, stats, (long)ND, 1);

    // ===== Layer 3: GATv2 (H=1,C=128, mean==identity) + graphLN  (-> d_out) =====
    f32_to_bf16<<<cvtBlocks, TPB, 0, stream>>>(buf1, Abf, (long)ND);
    gemm128_bf16_wmma<<<gemmBlocks, TPB, 0, stream>>>(Abf, wpkWl3, bl3, xlB, NN);
    gemm128_bf16_wmma<<<gemmBlocks, TPB, 0, stream>>>(Abf, wpkWr3, br3, xrB, NN);
    smax_init<<<(NN + TPB - 1) / TPB, TPB, 0, stream>>>((int*)smaxF, NN);
    hipMemsetAsync(denom, 0, (size_t)NN * sizeof(float), stream);
    gat_score <<<(int)((totEH1 + TPB - 1) / TPB), TPB, 0, stream>>>(xlB, xrB, at3, srcP, dstP,
                                                                    scores, (int*)smaxF, EE, NN, 1, DD);
    gat_expsum<<<(int)((totEH1 + TPB - 1) / TPB), TPB, 0, stream>>>(scores, dstP, (int*)smaxF,
                                                                    denom, EE, NN, 1);
    hipMemsetAsync(out, 0, ND * sizeof(float), stream);
    gat_agg<<<edgeWaveBl, TPB, 0, stream>>>(xlB, scores, denom, srcP, dstP, out, EE, NN, 1, DD);
    hipMemsetAsync(stats, 0, 2 * sizeof(float), stream);
    ln_stats<<<2048, TPB, 0, stream>>>(out, bg3, stats, (long)ND);
    ln_apply<<<(int)((ND + TPB - 1) / TPB), TPB, 0, stream>>>(out, bg3, g3, be3, stats, (long)ND, 0);
}
